// WindowAttention2D_63754494542400
// MI455X (gfx1250) — matrix operand order repaired and checked by reading, not compiled, not run on hardware
//
#include <hip/hip_runtime.h>

typedef __bf16 bf16;
typedef __attribute__((ext_vector_type(16))) __bf16 v16bf;
typedef __attribute__((ext_vector_type(8)))  __bf16 v8bf;
typedef __attribute__((ext_vector_type(4)))  __bf16 v4bf;
typedef __attribute__((ext_vector_type(2)))  __bf16 v2bf;
typedef __attribute__((ext_vector_type(8)))  float  v8f;
typedef __attribute__((ext_vector_type(4)))  float  v4f;
typedef int v2i32 __attribute__((vector_size(8)));  // matches builtin's pointee

#define AS1 __attribute__((address_space(1)))
#define AS3 __attribute__((address_space(3)))

#if __has_builtin(__builtin_amdgcn_global_load_async_to_lds_b64) && \
    __has_builtin(__builtin_amdgcn_s_wait_asynccnt)
#define HAVE_ASYNC_LDS 1
#else
#define HAVE_ASYNC_LDS 0
#endif

static __device__ __forceinline__ bf16 f2bf(float f) {
  unsigned u = __builtin_bit_cast(unsigned, f);
  unsigned r = u + 0x7FFFu + ((u >> 16) & 1u);  // round-to-nearest-even
  unsigned short hs = (unsigned short)(r >> 16);
  return __builtin_bit_cast(bf16, hs);
}

static __device__ __forceinline__ v2bf pk2bf(float x, float y) {
#if __has_builtin(__builtin_amdgcn_cvt_pk_bf16_f32)
  return __builtin_bit_cast(v2bf, __builtin_amdgcn_cvt_pk_bf16_f32(x, y));
#else
  return (v2bf){f2bf(x), f2bf(y)};
#endif
}

static __device__ __forceinline__ v4bf to4bf(v4f v) {
  v2bf lo = pk2bf(v.x, v.y), hi = pk2bf(v.z, v.w);
  return __builtin_shufflevector(lo, hi, 0, 1, 2, 3);
}
static __device__ __forceinline__ v4bf to4bf(v4bf v) { return v; }

static __device__ __forceinline__ v8f wmma_bf16f32(v16bf a, v16bf b, v8f c) {
  return __builtin_amdgcn_wmma_f32_16x16x32_bf16(false, a, false, b, (short)0, c,
                                                 false, false);
}

// A-fragment (16x32 bf16): lane half h: vgpr0-3 = K[h*8..+7], vgpr4-7 = K[16+h*8..+7]
static __device__ __forceinline__ v16bf load_a_frag(const bf16* rowp, int h) {
  v8bf lo = *(const v8bf*)(rowp + h * 8);
  v8bf hi = *(const v8bf*)(rowp + 16 + h * 8);
  return __builtin_shufflevector(lo, hi, 0, 1, 2, 3, 4, 5, 6, 7,
                                 8, 9, 10, 11, 12, 13, 14, 15);
}
// B-fragment (32x16 bf16): lane half h: vgpr0-3 = K[h*8..+7], vgpr4-7 = K[16+h*8..+7]
static __device__ __forceinline__ v16bf load_b_frag(const bf16* colp, int h) {
  v8bf lo = *(const v8bf*)(colp + h * 8);
  v8bf hi = *(const v8bf*)(colp + 16 + h * 8);
  return __builtin_shufflevector(lo, hi, 0, 1, 2, 3, 4, 5, 6, 7,
                                 8, 9, 10, 11, 12, 13, 14, 15);
}

template <typename T> struct avec;
template <> struct avec<float> { using t = v4f; };
template <> struct avec<bf16>  { using t = v4bf; };

template <typename AT>
static __device__ __forceinline__ void load_a_regs(const AT* __restrict__ A,
                                                   int Ktot, int m0, int kk,
                                                   int t,
                                                   typename avec<AT>::t ar[4]) {
  for (int r = 0; r < 4; ++r) {
    int lin = t + r * 256, row = lin >> 3, g = lin & 7;
    ar[r] = *(const typename avec<AT>::t*)(A + (size_t)(m0 + row) * Ktot + kk +
                                           g * 4);
  }
}
static __device__ __forceinline__ void load_b_regs(const float* __restrict__ B,
                                                   int Ntot, int n0, int kk,
                                                   int t, v4f br[4]) {
  for (int r = 0; r < 4; ++r) {
    int lin = t + r * 256, krow = lin >> 5, g = lin & 31;
    br[r] = *(const v4f*)(B + (size_t)(kk + krow) * Ntot + n0 + g * 4);
  }
}
template <typename AV>
static __device__ __forceinline__ void store_a_lds(bf16 (*Asb)[40], int t,
                                                   AV ar[4]) {
  for (int r = 0; r < 4; ++r) {
    int lin = t + r * 256, row = lin >> 3, g = lin & 7;
    *(v4bf*)&Asb[row][g * 4] = to4bf(ar[r]);
  }
}
static __device__ __forceinline__ void store_b_lds(bf16 (*Bsb)[40], int t,
                                                   v4f br[4]) {
  for (int r = 0; r < 4; ++r) {
    int lin = t + r * 256, krow = lin >> 5, g = lin & 31;
    Bsb[g * 4 + 0][krow] = f2bf(br[r].x);
    Bsb[g * 4 + 1][krow] = f2bf(br[r].y);
    Bsb[g * 4 + 2][krow] = f2bf(br[r].z);
    Bsb[g * 4 + 3][krow] = f2bf(br[r].w);
  }
}

// ---------------------------------------------------------------------------
// out[M x N] = A[M x K] * B[K x N] + bias. Double-buffered LDS pipeline,
// block tile 128x128x32, 256 thr = 8 waves (4M x 2N), wave tile 32x64 (2x4 WMMA)
// ---------------------------------------------------------------------------
template <typename AT, typename OT>
__global__ __launch_bounds__(256) void gemm_bias_kernel(
    const AT* __restrict__ A, const float* __restrict__ B,
    const float* __restrict__ bias, OT* __restrict__ out, int Ntot, int Ktot) {
  __shared__ bf16 As[2][128][40];  // [m][k], k-contiguous, padded
  __shared__ bf16 Bs[2][128][40];  // [n][k] (transposed), k-contiguous, padded

  const int t = threadIdx.x;
  const int wave = t >> 5, lane = t & 31;
  const int h = lane >> 4, l16 = lane & 15;
  const int wm = wave & 3, wn = wave >> 2;
  const int m0 = blockIdx.y * 128;
  const int n0 = blockIdx.x * 128;

  v8f acc[2][4];
  for (int mt = 0; mt < 2; ++mt)
    for (int nt = 0; nt < 4; ++nt)
      acc[mt][nt] = (v8f){0.f, 0.f, 0.f, 0.f, 0.f, 0.f, 0.f, 0.f};

  // Prologue: stage K-tile 0 into buffer 0
  {
    typename avec<AT>::t ar[4];
    v4f br[4];
    load_a_regs<AT>(A, Ktot, m0, 0, t, ar);
    load_b_regs(B, Ntot, n0, 0, t, br);
    store_a_lds(As[0], t, ar);
    store_b_lds(Bs[0], t, br);
  }

  const int nk = Ktot >> 5;
  for (int ki = 0; ki < nk; ++ki) {
    const int cur = ki & 1;
    const bool more = (ki + 1) < nk;
    // Issue next tile's global loads before the barrier (latency overlap)
    typename avec<AT>::t arn[4];
    v4f brn[4];
    if (more) {
      load_a_regs<AT>(A, Ktot, m0, (ki + 1) * 32, t, arn);
      load_b_regs(B, Ntot, n0, (ki + 1) * 32, t, brn);
    }
    __syncthreads();  // buffer `cur` ready; everyone done with `cur^1`

    v16bf af[2], bfv[4];
    for (int mt = 0; mt < 2; ++mt)
      af[mt] = load_a_frag(&As[cur][wm * 32 + mt * 16 + l16][0], h);
    for (int nt = 0; nt < 4; ++nt)
      bfv[nt] = load_b_frag(&Bs[cur][wn * 64 + nt * 16 + l16][0], h);
    for (int mt = 0; mt < 2; ++mt)
      for (int nt = 0; nt < 4; ++nt)
        acc[mt][nt] = wmma_bf16f32(af[mt], bfv[nt], acc[mt][nt]);

    if (more) {
      store_a_lds(As[cur ^ 1], t, arn);
      store_b_lds(Bs[cur ^ 1], t, brn);
    }
  }

  // Epilogue: bias add, store (C/D layout: vgpr r -> M = 8*h + r, N = l16)
  for (int mt = 0; mt < 2; ++mt) {
    for (int nt = 0; nt < 4; ++nt) {
      int ng = n0 + wn * 64 + nt * 16 + l16;
      float bv = bias[ng];
      for (int r = 0; r < 8; ++r) {
        int mg = m0 + wm * 32 + mt * 16 + h * 8 + r;
        float val = acc[mt][nt][r] + bv;
        if constexpr (__is_same(OT, float))
          out[(size_t)mg * Ntot + ng] = val;
        else
          out[(size_t)mg * Ntot + ng] = f2bf(val);
      }
    }
  }
}

// ---------------------------------------------------------------------------
// Windowed attention: one workgroup (4 waves, 128 thr) per (window, head).
// 49 tokens padded to 64; shift-roll folded into gather/scatter indices.
// qkv: [100352][768] bf16 (col = s*256 + head*32 + d); out: [100352][256] bf16
// ---------------------------------------------------------------------------
__global__ __launch_bounds__(128) void win_attn_kernel(
    const bf16* __restrict__ qkv, bf16* __restrict__ attn_out) {
  __shared__ bf16 Qs[64][40];   // [tok][d]
  __shared__ bf16 Ks[64][40];   // [tok][d]  (B of QK^T: K-contig = d)
  __shared__ bf16 Vs[32][72];   // [d][tok]  (B of P*V: K-contig = tok)
  __shared__ float Sf[64][68];  // scores fp32
  __shared__ bf16 Pm[64][72];   // softmax probs bf16
  __shared__ int ridx[64];

  const int t = threadIdx.x;
  const int wave = t >> 5, lane = t & 31;
  const int h = lane >> 4, l16 = lane & 15;

  const int head = blockIdx.x & 7;
  const int win = blockIdx.x >> 3;
  const int b = win >> 6;
  const int wyi = (win >> 3) & 7;
  const int wxi = win & 7;

  if (t < 64) {
    int p = t, r = 0;
    if (p < 49) {
      int wy = p / 7, wx = p % 7;
      int hh = (wyi * 7 + wy + 3) % 56;  // roll(-3) gather == roll(+3) scatter
      int ww = (wxi * 7 + wx + 3) % 56;
      r = b * 3136 + hh * 56 + ww;
    }
    ridx[p] = r;
  }
  // Zero the padded token rows/cols (49..63)
  const bf16 bz = __builtin_bit_cast(bf16, (unsigned short)0);
  for (int i = t; i < 480; i += 128) {
    int p = 49 + (i >> 5), c = i & 31;
    Qs[p][c] = bz;
    Ks[p][c] = bz;
    Vs[c][p] = bz;
  }
  __syncthreads();

  // Gather Q/K/V for the 49 valid tokens. Q/K rows are contiguous 8B chunks ->
  // async global->LDS copies when available; V needs a transpose -> manual.
  for (int job = t; job < 1536; job += 128) {
    int mat = job >> 9;  // 0=q 1=k 2=v
    int rem = job & 511;
    int p = rem >> 3, g = rem & 7;
    if (p < 49) {
      const bf16* src =
          qkv + (size_t)ridx[p] * 768 + mat * 256 + head * 32 + g * 4;
      if (mat == 2) {
        v4bf val = *(const v4bf*)src;
        for (int i = 0; i < 4; ++i) Vs[g * 4 + i][p] = val[i];
      } else {
        bf16* dst = (mat == 0) ? &Qs[p][g * 4] : &Ks[p][g * 4];
#if HAVE_ASYNC_LDS
        __builtin_amdgcn_global_load_async_to_lds_b64(
            (AS1 v2i32*)(bf16*)src, (AS3 v2i32*)dst, 0, 0);
#else
        *(v4bf*)dst = *(const v4bf*)src;
#endif
      }
    }
  }
#if HAVE_ASYNC_LDS
  __builtin_amdgcn_s_wait_asynccnt(0);
#endif
  __syncthreads();

  // S = Q * K^T : wave owns M-tile `wave`; 4 N-tiles, K=32 (one WMMA each)
  const v8f vzero = (v8f){0.f, 0.f, 0.f, 0.f, 0.f, 0.f, 0.f, 0.f};
  {
    v16bf qa = load_a_frag(&Qs[wave * 16 + l16][0], h);
    v8f s[4] = {vzero, vzero, vzero, vzero};
    for (int nt = 0; nt < 4; ++nt) {
      v16bf kb = load_b_frag(&Ks[nt * 16 + l16][0], h);
      s[nt] = wmma_bf16f32(qa, kb, s[nt]);
    }
    for (int nt = 0; nt < 4; ++nt)
      for (int r = 0; r < 8; ++r)
        Sf[wave * 16 + h * 8 + r][nt * 16 + l16] = s[nt][r];
  }
  __syncthreads();

  // Softmax per row over valid cols [0,49); Pm cols 49..63 forced to zero.
  if (t < 64) {
    const float scale = 0.17677669529663687f;  // 1/sqrt(32)
    float mx = -1e30f;
    for (int n = 0; n < 49; ++n) mx = fmaxf(mx, Sf[t][n] * scale);
    float sum = 0.f;
    for (int n = 0; n < 49; ++n) {
      float e = __expf(Sf[t][n] * scale - mx);
      Sf[t][n] = e;
      sum += e;
    }
    float inv = 1.f / sum;
    for (int n = 0; n < 48; n += 2)
      *(v2bf*)&Pm[t][n] = pk2bf(Sf[t][n] * inv, Sf[t][n + 1] * inv);
    Pm[t][48] = f2bf(Sf[t][48] * inv);
    for (int n = 49; n < 64; ++n) Pm[t][n] = bz;
  }
  __syncthreads();

  // O = P * V : K = 64 tokens (2 WMMA steps), N = 32 channels (2 N-tiles)
  v8f o[2] = {vzero, vzero};
  for (int kk = 0; kk < 64; kk += 32) {
    v16bf pa = load_a_frag(&Pm[wave * 16 + l16][kk], h);
    for (int nt = 0; nt < 2; ++nt) {
      v16bf vb = load_b_frag(&Vs[nt * 16 + l16][kk], h);
      o[nt] = wmma_bf16f32(pa, vb, o[nt]);
    }
  }

  // Scatter to original token order (un-window + un-roll fused)
  for (int nt = 0; nt < 2; ++nt) {
    int ch = head * 32 + nt * 16 + l16;
    for (int r = 0; r < 8; ++r) {
      int p = wave * 16 + h * 8 + r;
      if (p < 49) attn_out[(size_t)ridx[p] * 256 + ch] = f2bf(o[nt][r]);
    }
  }
}

// ---------------------------------------------------------------------------
extern "C" void kernel_launch(void* const* d_in, const int* in_sizes, int n_in,
                              void* d_out, int out_size, void* d_ws,
                              size_t ws_size, hipStream_t stream) {
  const float* x = (const float*)d_in[0];       // [100352, 256]
  const float* w_qkv = (const float*)d_in[1];   // [256, 768]
  const float* b_qkv = (const float*)d_in[2];   // [768]
  const float* w_proj = (const float*)d_in[3];  // [256, 256]
  const float* b_proj = (const float*)d_in[4];  // [256]
  (void)in_sizes; (void)n_in; (void)out_size; (void)ws_size;

  const int M = 32 * 56 * 56;  // 100352 tokens
  bf16* qkv_ws = (bf16*)d_ws;                                  // [M][768] bf16
  bf16* attn_ws = (bf16*)((char*)d_ws + (size_t)M * 768 * 2);  // [M][256] bf16
  float* out = (float*)d_out;

  // 1) QKV projection: M=784*128, N=6*128, K=256
  gemm_bias_kernel<float, bf16>
      <<<dim3(6, 784), 256, 0, stream>>>(x, w_qkv, b_qkv, qkv_ws, 768, 256);

  // 2) Shifted-window attention: 2048 windows * 8 heads
  win_attn_kernel<<<dim3(2048 * 8), 128, 0, stream>>>(qkv_ws, attn_ws);

  // 3) Output projection: N=2*128
  gemm_bias_kernel<bf16, float>
      <<<dim3(2, 784), 256, 0, stream>>>(attn_ws, w_proj, b_proj, out, 256, 256);
}
